// SparseUncompressConvolution_44349832299007
// MI455X (gfx1250) — matrix operand order repaired and checked by reading, not compiled, not run on hardware
//
#include <hip/hip_runtime.h>

// ---------------------------------------------------------------------------
// SparseUncompressConvolution on gfx1250 (MI455X).
//
// Math (derived from reference):
//   even rows w=2j  : out = lq + relu(U[j]@W1^T + b) + U[j]/3
//   odd  rows w=2j+1: out = lq + relu(U[j]@W0^T + U[j+1]@W2^T + b) + (U[j]+U[j+1])/3
// Three 512x512 GEMMs over 32768 rows -> ~51.5 GFLOP, ~0.4 GB traffic.
// Memory-bound at 23.3 TB/s (~15-20us); bf16 WMMA makes compute negligible.
// This revision: explicit software pipeline over 8 (kh,nt) groups with
// double-buffered B fragments and __builtin_amdgcn_sched_barrier(0) fences so
// group g+1's ds_loads are issued before group g's WMMAs (breaks the
// load -> s_wait_dscnt 0 -> wmma serialization the scheduler kept recreating).
// ---------------------------------------------------------------------------

#define B_      4
#define H_      8
#define N_      2048
#define NC_     1024
#define D_      512
#define W3_     (3 * D_)          // 1536, W row stride

#define M_TILE  128               // compressed rows per workgroup
#define N_TILE  64                // output channels per workgroup
#define K_TILE  64                // reduction chunk per staging round (2 WMMA-K)
#define U_ROWS  (M_TILE + 1)      // +1 row for the U[j+1] shift
#define U_STRIDE 72               // ushorts per LDS U row (64 + pad, 144B: 16B aligned)
#define W_STRIDE 72               // ushorts per LDS W K-row (64 + pad, 144B: 16B aligned)
#define W_MAT    (K_TILE * W_STRIDE)

typedef __attribute__((ext_vector_type(16))) __bf16 bf16x16;
typedef __attribute__((ext_vector_type(8)))  __bf16 bf16x8;
typedef __attribute__((ext_vector_type(8)))  float  f32x8;

union AB16 { bf16x16 v; bf16x8 h[2]; };

// two f32 -> packed bf16 pair (native cvt)
__device__ __forceinline__ unsigned pkbf(float lo, float hi) {
  unsigned short a = __builtin_bit_cast(unsigned short, (__bf16)lo);
  unsigned short b = __builtin_bit_cast(unsigned short, (__bf16)hi);
  return (unsigned)a | ((unsigned)b << 16);
}

__global__ __launch_bounds__(256)
void suc_wmma_kernel(const float* __restrict__ lq,
                     const float* __restrict__ uin,
                     const float* __restrict__ W,
                     const float* __restrict__ bias,
                     float* __restrict__ out) {
  __shared__ unsigned short ldsU[U_ROWS * U_STRIDE];      // 18.1 KB
  __shared__ unsigned short ldsW[3 * W_MAT];              // 27.0 KB

  const int tid  = threadIdx.x;
  const int lane = tid & 31;
  const int wv   = tid >> 5;          // wave id 0..7 -> 16-row m-subtile
  const int m    = lane & 15;         // M / N lane index within 16x16 tile
  const int hi   = lane >> 4;         // half-wave select (K split / M split)

  const int n0 = blockIdx.x * N_TILE; // output-channel tile
  const int j0 = blockIdx.y * M_TILE; // compressed-row tile
  const int bh = blockIdx.z;          // 0..31

  const long uBase  = (long)bh * NC_ * D_;
  const long lqBase = (long)bh * N_  * D_;

  f32x8 accE[4] = {};                 // even rows: A0 x W1
  f32x8 accO[4] = {};                 // odd rows:  A0 x W0 + A1 x W2

  // fragment load helpers (kh = K half of the staged tile, nt = 16-col N tile)
  const unsigned short* urow = ldsU + (wv * 16 + m) * U_STRIDE + hi * 8;
  auto loadB = [&](int g, AB16* bb) {
    const int kh = g >> 2, nt = g & 3;
    const unsigned short* wb = ldsW + (nt * 16 + m) * W_STRIDE + kh * 32 + hi * 8;
    bb[0].h[0] = *(const bf16x8*)(wb);
    bb[0].h[1] = *(const bf16x8*)(wb + 16);
    bb[1].h[0] = *(const bf16x8*)(wb + 1 * W_MAT);
    bb[1].h[1] = *(const bf16x8*)(wb + 1 * W_MAT + 16);
    bb[2].h[0] = *(const bf16x8*)(wb + 2 * W_MAT);
    bb[2].h[1] = *(const bf16x8*)(wb + 2 * W_MAT + 16);
  };

  for (int k0 = 0; k0 < D_; k0 += K_TILE) {
    // ---- stage U tile (129 rows x 64 K) as packed bf16 pairs ----
    {
      const int c2 = (tid & 31) * 2;
      for (int r = (tid >> 5); r < U_ROWS; r += 8) {
        const int row = j0 + r;
        float2 v = make_float2(0.0f, 0.0f);
        if (row < NC_)
          v = *(const float2*)(uin + uBase + (long)row * D_ + k0 + c2);
        *(unsigned*)&ldsU[r * U_STRIDE + c2] = pkbf(v.x, v.y);
      }
    }
    // ---- stage W0/W1/W2 N-major: ldsW[w][nn][kk] ----
    for (int idx = tid; idx < 3 * K_TILE * (N_TILE / 2); idx += 256) {
      const int w   = idx >> 11;            // / (64*32)
      const int rem = idx & 2047;
      const int nn  = rem & 63;
      const int kk2 = (rem >> 6) * 2;
      const float* wp = W + (long)(n0 + nn) * W3_ + w * D_ + k0 + kk2;
      const float vlo = wp[0];
      const float vhi = wp[1];
      *(unsigned*)&ldsW[w * W_MAT + nn * W_STRIDE + kk2] = pkbf(vlo, vhi);
    }
    // prefetch next K-tile of U into L2 (global_prefetch_b8)
    if (k0 + K_TILE < D_) {
      const int prow = j0 + (tid >> 5);
      if (prow < NC_)
        __builtin_prefetch(&uin[uBase + (long)prow * D_ + k0 + K_TILE + (tid & 31) * 2], 0, 1);
    }
    __syncthreads();

    // ---- software-pipelined 24 WMMAs per wave per staging round ----
    // Preload A fragments for both K halves (rows wv*16+m and +1, ISA layout).
    AB16 a0[2], a1[2];
#pragma unroll
    for (int kh = 0; kh < 2; ++kh) {
      const unsigned short* ur = urow + kh * 32;
      a0[kh].h[0] = *(const bf16x8*)(ur);
      a0[kh].h[1] = *(const bf16x8*)(ur + 16);
      ur += U_STRIDE;
      a1[kh].h[0] = *(const bf16x8*)(ur);
      a1[kh].h[1] = *(const bf16x8*)(ur + 16);
    }
    AB16 bb[2][3];                      // double-buffered B fragments
    loadB(0, bb[0]);
#pragma unroll
    for (int g = 0; g < 8; ++g) {
      if (g + 1 < 8) loadB(g + 1, bb[(g + 1) & 1]);   // prefetch next group
      __builtin_amdgcn_sched_barrier(0);              // pin loads above WMMAs
      const AB16* Bc = bb[g & 1];
      const int kh = g >> 2, nt = g & 3;
      accE[nt] = __builtin_amdgcn_wmma_f32_16x16x32_bf16(
          false, a0[kh].v, false, Bc[1].v, (short)0, accE[nt], false, false);
      accO[nt] = __builtin_amdgcn_wmma_f32_16x16x32_bf16(
          false, a0[kh].v, false, Bc[0].v, (short)0, accO[nt], false, false);
      accO[nt] = __builtin_amdgcn_wmma_f32_16x16x32_bf16(
          false, a1[kh].v, false, Bc[2].v, (short)0, accO[nt], false, false);
    }
    __syncthreads();
  }

  // ---- fused epilogue: bias + ReLU + residual (/3) + skip connection ----
  // lq/out are streamed exactly once -> non-temporal; uin stays cacheable.
  const float inv3 = 1.0f / 3.0f;
  const int jbase = j0 + wv * 16 + 8 * hi;   // D frag: M = v + 8*(lane/16)
#pragma unroll
  for (int nt = 0; nt < 4; ++nt) {
    const int d = n0 + nt * 16 + m;          // output channel (D frag: N = lane%16)
    const float bsv = bias[d];
    const float* ucol = uin + uBase + d;
    float uu[9];
#pragma unroll
    for (int v = 0; v < 9; ++v) {            // 9 rows give both U[j] and U[j+1]
      const int row = jbase + v;
      uu[v] = (row < NC_) ? ucol[(long)row * D_] : 0.0f;
    }
#pragma unroll
    for (int v = 0; v < 8; ++v) {
      const int j = jbase + v;
      const long oE = lqBase + (long)(2 * j) * D_ + d;     // even row w = 2j
      const long oO = oE + D_;                             // odd  row w = 2j+1
      const float e = fmaxf(accE[nt][v] + bsv, 0.0f) + uu[v] * inv3;
      const float o = fmaxf(accO[nt][v] + bsv, 0.0f) + (uu[v] + uu[v + 1]) * inv3;
      __builtin_nontemporal_store(__builtin_nontemporal_load(lq + oE) + e, out + oE);
      __builtin_nontemporal_store(__builtin_nontemporal_load(lq + oO) + o, out + oO);
    }
  }
}

extern "C" void kernel_launch(void* const* d_in, const int* in_sizes, int n_in,
                              void* d_out, int out_size, void* d_ws, size_t ws_size,
                              hipStream_t stream) {
  (void)in_sizes; (void)n_in; (void)d_ws; (void)ws_size; (void)out_size;
  const float* lq   = (const float*)d_in[0];  // (4,8,2048,512)
  const float* uin  = (const float*)d_in[1];  // (4,8,1024,512)
  const float* W    = (const float*)d_in[2];  // (512,1536)
  const float* bias = (const float*)d_in[3];  // (512,)
  // d_in[4] = i (unused by the reference math)
  float* out = (float*)d_out;                 // (4,8,2048,512)

  dim3 grid(D_ / N_TILE, NC_ / M_TILE, B_ * H_);  // (8, 8, 32)
  dim3 block(256);
  suc_wmma_kernel<<<grid, block, 0, stream>>>(lq, uin, W, bias, out);
}
